// GNNModel_12661563589030
// MI455X (gfx1250) — compile-verified, matrix operand order checked by reading, not run on hardware
//
#include <hip/hip_runtime.h>

typedef __attribute__((ext_vector_type(16))) _Float16 v16h;
typedef __attribute__((ext_vector_type(8)))  _Float16 v8h;
typedef __attribute__((ext_vector_type(4)))  _Float16 v4h;
typedef __attribute__((ext_vector_type(8)))  float    v8f;

static constexpr int   NNODES = 50000;
static constexpr float SLOPE  = 0.2f;

// ---- order-preserving int encoding of float (for atomicMax segment-max) ----
__device__ __forceinline__ int fenc(float f) {
    int b = __float_as_int(f);
    return b >= 0 ? b : (int)(0x80000000u - (unsigned)b);
}
__device__ __forceinline__ float fdec(int e) {
    return e >= 0 ? __int_as_float(e) : __uint_as_float(0x80000000u - (unsigned)e);
}

// ------------------------------------------------------------------
// f32 -> packed f16 conversion, 8 elements / thread
// ------------------------------------------------------------------
__global__ void conv_f16(const float* __restrict__ in, _Float16* __restrict__ out, int total8) {
    int t = blockIdx.x * blockDim.x + threadIdx.x;
    if (t >= total8) return;
    const float4* i4 = (const float4*)(in + t * 8);
    float4 a = i4[0], b = i4[1];
    v8h o;
    o[0]=(_Float16)a.x; o[1]=(_Float16)a.y; o[2]=(_Float16)a.z; o[3]=(_Float16)a.w;
    o[4]=(_Float16)b.x; o[5]=(_Float16)b.y; o[6]=(_Float16)b.z; o[7]=(_Float16)b.w;
    *(v8h*)(out + t * 8) = o;
}

// ------------------------------------------------------------------
// Dual GEMM:  XL = Xh @ Wl + bl ,  XR = Xh @ Wr + br   (Xh is f16)
//
// grid.y = tn (16-col tile of W), grid.x covers tm in groups of 8.
// Block (256 thr = 8 waves) stages the K x 16 tiles of Wl and Wr into
// LDS once, pre-swizzled into the WMMA B-fragment layout (each lane's
// 16 halves contiguous -> 2x ds_load_b128). One wave per 16x16 C tile,
// v_wmma_f32_16x16x32_f16, A fragment = two global_load_b128 (f16),
// shared by both accumulator chains.
// ------------------------------------------------------------------
template <int K, int DOUT>
__global__ __launch_bounds__(256)
void gemm_dual_wmma(const _Float16* __restrict__ Xh,
                    const float* __restrict__ Wl, const float* __restrict__ Wr,
                    const float* __restrict__ bl, const float* __restrict__ br,
                    float* __restrict__ XL, float* __restrict__ XR,
                    int tmTiles) {
    constexpr int KSTEPS = K / 32;
    __shared__ _Float16 sBL[K * 16];
    __shared__ _Float16 sBR[K * 16];

    const int tn  = blockIdx.y;
    const int tid = threadIdx.x;

    // ---- stage W tiles (bounds-masked here, NOT in the hot loop) ----
    for (int idx = tid; idx < K * 16; idx += 256) {
        int k = idx >> 4;
        int n = idx & 15;
        int col = tn * 16 + n;
        float vl = 0.f, vr = 0.f;
        if (col < DOUT) {
            vl = Wl[k * DOUT + col];
            vr = Wr[k * DOUT + col];
        }
        int r    = k & 31;
        int s    = k >> 5;
        int lane = ((r >> 3) & 1) * 16 + n;
        int j    = (r & 7) + ((r & 16) ? 8 : 0);
        int slot = (s * 32 + lane) * 16 + j;
        sBL[slot] = (_Float16)vl;
        sBR[slot] = (_Float16)vr;
    }
    __syncthreads();

    const int wv = tid >> 5;
    const int tm = blockIdx.x * 8 + wv;
    if (tm >= tmTiles) return;               // wave-uniform

    const int lane  = tid & 31;
    const int l15   = lane & 15;
    const int khalf = (lane >> 4) * 8;
    const int row   = tm * 16 + l15;         // 50000 = 3125*16: always valid
    const int col   = tn * 16 + l15;

    const _Float16* Xrow = Xh + row * K;
    v8f accL = {}; v8f accR = {};

#pragma unroll
    for (int s = 0; s < KSTEPS; ++s) {
        const int k0 = s * 32;
        // ---- A fragment: two 16B f16 loads, no conversion ----
        v8h r0 = *(const v8h*)(Xrow + k0 + khalf);
        v8h r1 = *(const v8h*)(Xrow + k0 + khalf + 16);
        v16h a;
#pragma unroll
        for (int j = 0; j < 8; ++j) { a[j] = r0[j]; a[8 + j] = r1[j]; }

        // ---- B fragments: one contiguous 32B LDS read per matrix ----
        const int fbase = (s * 32 + lane) * 16;
        v16h bLv = *(const v16h*)(&sBL[fbase]);
        v16h bRv = *(const v16h*)(&sBR[fbase]);

        accL = __builtin_amdgcn_wmma_f32_16x16x32_f16(false, a, false, bLv, (short)0, accL, false, false);
        accR = __builtin_amdgcn_wmma_f32_16x16x32_f16(false, a, false, bRv, (short)0, accR, false, false);
    }

    if (col < DOUT) {
        float addL = bl[col], addR = br[col];
        int mbase = tm * 16 + (lane >> 4) * 8;   // C layout: vgpr i -> M = mbase+i, N = col
#pragma unroll
        for (int i = 0; i < 8; ++i) {
            XL[(mbase + i) * DOUT + col] = accL[i] + addL;
            XR[(mbase + i) * DOUT + col] = accR[i] + addR;
        }
    }
}

// ------------------------------------------------------------------
// Per-layer init: zero output accumulator + denom, menc = INT_MIN
// (every node has a self loop, so INT_MIN never survives the max)
// ------------------------------------------------------------------
__global__ void init_layer(float* __restrict__ H, int nH,
                           float* __restrict__ denom, int* __restrict__ menc, int N) {
    int t = blockIdx.x * blockDim.x + threadIdx.x;
    if (t < nH) H[t] = 0.f;
    if (t < N) { denom[t] = 0.f; menc[t] = (int)0x80000000; }
}

// ------------------------------------------------------------------
// Edge pass 1: logits[e] = leaky_relu(xl[src]+xr[dst]) . att ; seg-max
// ------------------------------------------------------------------
template <int D>
__global__ void edge_logits(const int* __restrict__ ei, int E, int N,
                            const float* __restrict__ XL, const float* __restrict__ XR,
                            const float* __restrict__ att,
                            float* __restrict__ logits, int* __restrict__ menc) {
    int e = blockIdx.x * blockDim.x + threadIdx.x;
    int ET = E + N;
    if (e >= ET) return;
    int src = e < E ? ei[e]     : e - E;
    int dst = e < E ? ei[E + e] : e - E;
    const float4* l  = (const float4*)(XL + src * D);
    const float4* r  = (const float4*)(XR + dst * D);
    const float4* a4 = (const float4*)att;
    float s = 0.f;
#pragma unroll
    for (int i = 0; i < (D >> 2); ++i) {
        float4 lv = l[i], rv = r[i], av = a4[i];
        float v;
        v = lv.x + rv.x; v = v > 0.f ? v : SLOPE * v; s += v * av.x;
        v = lv.y + rv.y; v = v > 0.f ? v : SLOPE * v; s += v * av.y;
        v = lv.z + rv.z; v = v > 0.f ? v : SLOPE * v; s += v * av.z;
        v = lv.w + rv.w; v = v > 0.f ? v : SLOPE * v; s += v * av.w;
    }
    logits[e] = s;
    atomicMax(menc + dst, fenc(s));
}

// ------------------------------------------------------------------
// Edge pass 2: ex = exp(logit - max[dst]);  denom[dst] += ex
// ------------------------------------------------------------------
__global__ void edge_exp(const int* __restrict__ ei, int E, int N,
                         const float* __restrict__ logits, const int* __restrict__ menc,
                         float* __restrict__ exb, float* __restrict__ denom) {
    int e = blockIdx.x * blockDim.x + threadIdx.x;
    int ET = E + N;
    if (e >= ET) return;
    int dst = e < E ? ei[E + e] : e - E;
    float ex = __expf(logits[e] - fdec(menc[dst]));
    exb[e] = ex;
    unsafeAtomicAdd(denom + dst, ex);
}

// ------------------------------------------------------------------
// Edge pass 3: H[dst] += (ex/denom[dst]) * xl[src]
// thread = (edge, 4-feature chunk) -> coalesced loads & atomics
// ------------------------------------------------------------------
template <int D>
__global__ void edge_scatter(const int* __restrict__ ei, int E, int N,
                             const float* __restrict__ XL,
                             const float* __restrict__ exb, const float* __restrict__ denom,
                             float* __restrict__ H, int total) {
    int t = blockIdx.x * blockDim.x + threadIdx.x;
    if (t >= total) return;
    constexpr int CH = D >> 2;
    int e = t / CH;
    int c = t % CH;
    int src = e < E ? ei[e]     : e - E;
    int dst = e < E ? ei[E + e] : e - E;
    float alpha = exb[e] / denom[dst];
    float4 x = ((const float4*)(XL + src * D))[c];
    float* o = H + dst * D + c * 4;
    unsafeAtomicAdd(o + 0, alpha * x.x);
    unsafeAtomicAdd(o + 1, alpha * x.y);
    unsafeAtomicAdd(o + 2, alpha * x.z);
    unsafeAtomicAdd(o + 3, alpha * x.w);
}

// ------------------------------------------------------------------
// bias + ReLU, emitting the f16 input for the next layer's GEMM.
// 4 elements / thread, packed 8B f16 store.
// ------------------------------------------------------------------
template <int D>
__global__ void bias_relu_f16(const float* __restrict__ H, const float* __restrict__ c,
                              _Float16* __restrict__ Hh, int total4) {
    int t = blockIdx.x * blockDim.x + threadIdx.x;
    if (t >= total4) return;
    float4 v = ((const float4*)H)[t];
    const float4* c4 = (const float4*)c;
    float4 cb = c4[t & ((D >> 2) - 1)];          // D is a power of two here (128, 64)
    float a0 = v.x + cb.x, a1 = v.y + cb.y, a2 = v.z + cb.z, a3 = v.w + cb.w;
    v4h o;
    o[0] = (_Float16)(a0 > 0.f ? a0 : 0.f);
    o[1] = (_Float16)(a1 > 0.f ? a1 : 0.f);
    o[2] = (_Float16)(a2 > 0.f ? a2 : 0.f);
    o[3] = (_Float16)(a3 > 0.f ? a3 : 0.f);
    *(v4h*)(Hh + t * 4) = o;
}

// ------------------------------------------------------------------
// final: out = log_softmax(H + c) per row (D = 40)
// ------------------------------------------------------------------
__global__ void log_softmax_k(const float* __restrict__ H, const float* __restrict__ c,
                              float* __restrict__ out, int N) {
    constexpr int D = 40;
    int n = blockIdx.x * blockDim.x + threadIdx.x;
    if (n >= N) return;
    const float* h = H + n * D;
    float buf[D];
    float mx = -3.4e38f;
#pragma unroll
    for (int i = 0; i < D; ++i) { buf[i] = h[i] + c[i]; mx = fmaxf(mx, buf[i]); }
    float s = 0.f;
#pragma unroll
    for (int i = 0; i < D; ++i) s += __expf(buf[i] - mx);
    float ls = __logf(s) + mx;
    float* o = out + n * D;
#pragma unroll
    for (int i = 0; i < D; ++i) o[i] = buf[i] - ls;
}

// ------------------------------------------------------------------
// per-layer driver (host side)
// ------------------------------------------------------------------
template <int DI, int DOUT>
static void run_layer(const _Float16* XinH, const float* Wl, const float* Wr,
                      const float* bl, const float* br, const float* att, const float* cb,
                      float* XL, float* XR, float* Hout, _Float16* HhNext,
                      float* logits, float* exb, float* denom, int* menc,
                      const int* ei, int E, int N, bool relu,
                      float* finalOut, hipStream_t stream) {
    constexpr int TILES_N = (DOUT + 15) / 16;
    const int tmTiles = N / 16;                         // 3125
    dim3 grid((tmTiles + 7) / 8, TILES_N);
    gemm_dual_wmma<DI, DOUT><<<grid, 256, 0, stream>>>(XinH, Wl, Wr, bl, br, XL, XR, tmTiles);

    const int nH = N * DOUT;
    init_layer<<<(nH + 255) / 256, 256, 0, stream>>>(Hout, nH, denom, menc, N);

    const int ET = E + N;
    edge_logits<DOUT><<<(ET + 255) / 256, 256, 0, stream>>>(ei, E, N, XL, XR, att, logits, menc);
    edge_exp<<<(ET + 255) / 256, 256, 0, stream>>>(ei, E, N, logits, menc, exb, denom);
    const int total = ET * (DOUT / 4);
    edge_scatter<DOUT><<<(total + 255) / 256, 256, 0, stream>>>(ei, E, N, XL, exb, denom, Hout, total);

    if (relu) {
        const int total4 = nH / 4;
        bias_relu_f16<DOUT><<<(total4 + 255) / 256, 256, 0, stream>>>(Hout, cb, HhNext, total4);
    } else {
        log_softmax_k<<<(N + 255) / 256, 256, 0, stream>>>(Hout, cb, finalOut, N);
    }
}

// ------------------------------------------------------------------
extern "C" void kernel_launch(void* const* d_in, const int* in_sizes, int n_in,
                              void* d_out, int out_size, void* d_ws, size_t ws_size,
                              hipStream_t stream) {
    (void)n_in; (void)out_size; (void)ws_size;
    const float* x  = (const float*)d_in[0];
    const int*   ei = (const int*)d_in[1];
    const int E  = in_sizes[1] / 2;
    const int N  = NNODES;
    const int ET = E + N;

    float* ws = (float*)d_ws;
    size_t off = 0;
    float* XL     = ws + off; off += (size_t)N * 128;
    float* XR     = ws + off; off += (size_t)N * 128;
    float* H1     = ws + off; off += (size_t)N * 128;
    float* H2     = ws + off; off += (size_t)N * 64;
    float* H3     = ws + off; off += (size_t)N * 40;
    float* logits = ws + off; off += (size_t)ET;
    float* exb    = ws + off; off += (size_t)ET;
    float* denom  = ws + off; off += (size_t)N;
    int*   menc   = (int*)(ws + off); off += (size_t)N;
    _Float16* Xh  = (_Float16*)(ws + off); off += (size_t)N * 128;  // N*256 halves

    const float** p = (const float**)d_in;

    // x (f32) -> Xh (f16), once
    const int total8 = N * 256 / 8;
    conv_f16<<<(total8 + 255) / 256, 256, 0, stream>>>(x, Xh, total8);

    // layer epilogues overwrite Xh with the next layer's f16 input
    run_layer<256, 128>(Xh, p[2],  p[3],  p[4],  p[5],  p[6],  p[7],
                        XL, XR, H1, Xh, logits, exb, denom, menc, ei, E, N, true,  nullptr, stream);
    run_layer<128, 64>(Xh, p[8],  p[9],  p[10], p[11], p[12], p[13],
                        XL, XR, H2, Xh, logits, exb, denom, menc, ei, E, N, true,  nullptr, stream);
    run_layer<64, 40>(Xh, p[14], p[15], p[16], p[17], p[18], p[19],
                        XL, XR, H3, nullptr, logits, exb, denom, menc, ei, E, N, false, (float*)d_out, stream);
}
